// SHGNN_43061342110478
// MI455X (gfx1250) — matrix operand order repaired and checked by reading, not compiled, not run on hardware
//
#include <hip/hip_runtime.h>

typedef __attribute__((ext_vector_type(16))) __bf16 v16bf;
typedef __attribute__((ext_vector_type(8)))  __bf16 v8bf;
typedef __attribute__((ext_vector_type(8)))  float  v8f;

#define N_NODES 50000
#define N_EDGES 20000
#define M_INC   400000
#define L_EDG   800000
#define HID     128
#define BATCH_B 32
#define NCLS    10

__device__ __forceinline__ void gadd(float* p, float v) {
    __hip_atomic_fetch_add(p, v, __ATOMIC_RELAXED, __HIP_MEMORY_SCOPE_AGENT);
}
__device__ __forceinline__ void ladd(float* p, float v) {
    __hip_atomic_fetch_add(p, v, __ATOMIC_RELAXED, __HIP_MEMORY_SCOPE_WORKGROUP);
}

// Convert We (q==0) and 16 Wg matrices (q>=1) f32 [k,n] -> bf16 transposed [n,k].
__global__ void wconv_kernel(const float* __restrict__ We, const float* __restrict__ Wg,
                             __bf16* __restrict__ out) {
    int t = blockIdx.x * 256 + threadIdx.x;
    if (t >= 17 * 16384) return;
    int q = t >> 14, o = t & 16383;
    int n = o >> 7, k = o & 127;
    const float* src = (q == 0) ? We : (Wg + (size_t)(q - 1) * 16384);
    out[(size_t)q * 16384 + (size_t)n * 128 + k] = (__bf16)src[(size_t)k * 128 + n];
}

// A_bf[m,:] = (bf16) src_f32[idx[m],:]    (thread = (m, 8 cols))
__global__ void gather_to_bf(__bf16* __restrict__ dst, const float* __restrict__ src,
                             const int* __restrict__ idx, int M) {
    long t = blockIdx.x * 256L + threadIdx.x;
    if (t >= (long)M * 16) return;
    int m = (int)(t >> 4), c = (int)(t & 15) << 3;
    const float* s = src + (size_t)idx[m] * HID + c;
    float4 a = *(const float4*)(s);
    float4 b = *(const float4*)(s + 4);
    v8bf o;
    o[0]=(__bf16)a.x; o[1]=(__bf16)a.y; o[2]=(__bf16)a.z; o[3]=(__bf16)a.w;
    o[4]=(__bf16)b.x; o[5]=(__bf16)b.y; o[6]=(__bf16)b.z; o[7]=(__bf16)b.w;
    *(v8bf*)(dst + (size_t)m * HID + c) = o;
}

// G[m,:] = (1+eps) * x_bf[m,:]   (f32 accumulator init; replaces memset)
__global__ void ginit_kernel(float* __restrict__ G, const __bf16* __restrict__ x,
                             const float* __restrict__ epsP, int M) {
    long t = blockIdx.x * 256L + threadIdx.x;
    if (t >= (long)M * 16) return;
    float alpha = 1.0f + epsP[0];
    int m = (int)(t >> 4), c = (int)(t & 15) << 3;
    v8bf v = *(const v8bf*)(x + (size_t)m * HID + c);
    float* g = G + (size_t)m * HID + c;
    float4 r0, r1;
    r0.x = alpha * (float)v[0]; r0.y = alpha * (float)v[1];
    r0.z = alpha * (float)v[2]; r0.w = alpha * (float)v[3];
    r1.x = alpha * (float)v[4]; r1.y = alpha * (float)v[5];
    r1.z = alpha * (float)v[6]; r1.w = alpha * (float)v[7];
    *(float4*)(g) = r0; *(float4*)(g + 4) = r1;
}

// G[de[e],:] += x_bf[se[e],:]   (inner-graph segment_sum; bf16 gather, f32 atomics)
__global__ void scatter_edges_bf(float* __restrict__ G, const __bf16* __restrict__ x,
                                 const int* __restrict__ se, const int* __restrict__ de, int nE) {
    long t = blockIdx.x * 256L + threadIdx.x;
    if (t >= (long)nE * 16) return;
    int e = (int)(t >> 4), c = (int)(t & 15) << 3;
    v8bf v = *(const v8bf*)(x + (size_t)se[e] * HID + c);
    float* p = G + (size_t)de[e] * HID + c;
#pragma unroll
    for (int j = 0; j < 8; ++j) gadd(p + j, (float)v[j]);
}

// dst_f32[seg[m],:] += x_bf[m,:]   (pooling segment_sum)
__global__ void pool_rows_bf(float* __restrict__ dst, const __bf16* __restrict__ x,
                             const int* __restrict__ seg, int M) {
    long t = blockIdx.x * 256L + threadIdx.x;
    if (t >= (long)M * 16) return;
    int m = (int)(t >> 4), c = (int)(t & 15) << 3;
    v8bf v = *(const v8bf*)(x + (size_t)m * HID + c);
    float* p = dst + (size_t)seg[m] * HID + c;
#pragma unroll
    for (int j = 0; j < 8; ++j) gadd(p + j, (float)v[j]);
}

__global__ void relu_ip(float* __restrict__ x, long n) {
    long t = blockIdx.x * 256L + threadIdx.x;
    if (t < n) { float v = x[t]; x[t] = v > 0.f ? v : 0.f; }
}

// pooled[batch[m],:] += x_f32[oni[m],:]   (LDS-staged, low-contention)
__global__ void pool_batch(float* __restrict__ pooled, const float* __restrict__ x,
                           const int* __restrict__ oni, const int* __restrict__ bat, int M) {
    __shared__ float acc[BATCH_B * HID];
    for (int i = threadIdx.x; i < BATCH_B * HID; i += 256) acc[i] = 0.f;
    __syncthreads();
    long total = (long)M * 32;
    for (long t = blockIdx.x * 256L + threadIdx.x; t < total; t += (long)gridDim.x * 256L) {
        int m = (int)(t >> 5), c = (int)(t & 31) << 2;
        float4 v = *(const float4*)(x + (size_t)oni[m] * HID + c);
        float* p = acc + bat[m] * HID + c;
        ladd(p + 0, v.x); ladd(p + 1, v.y); ladd(p + 2, v.z); ladd(p + 3, v.w);
    }
    __syncthreads();
    for (int i = threadIdx.x; i < BATCH_B * HID; i += 256) gadd(&pooled[i], acc[i]);
}

// score[b,c] += bp[c] + sum_k pooled[b,k]*Wp[k,c]
__global__ void head_kernel(float* __restrict__ score, const float* __restrict__ pooled,
                            const float* __restrict__ Wp, const float* __restrict__ bp) {
    int t = threadIdx.x;
    if (t >= BATCH_B * NCLS) return;
    int b = t / NCLS, c = t % NCLS;
    float s = bp[c];
    for (int k = 0; k < HID; ++k) s += pooled[(size_t)b * HID + k] * Wp[(size_t)k * NCLS + c];
    score[t] += s;
}

// Y[rows,128] = post( X @ W + bias ); X f32 or bf16, Y f32 or bf16, optional eval-BN, optional ReLU.
// Wt is bf16 transposed: Wt[n*128 + k] = W[k,n]. bf16 WMMA, f32 accumulate.
// One wave = 16x128 output strip: 8 tiles of 16x16, K = 4 chunks of 32.
template <int IN_BF, int OUT_BF, int USE_BN>
__global__ __launch_bounds__(256) void gemm128(
    const void* __restrict__ Xv, const __bf16* __restrict__ Wt,
    const float* __restrict__ bias, const float* __restrict__ gamma,
    const float* __restrict__ beta, void* __restrict__ Yv,
    int rows, int doRelu)
{
    const int lane = threadIdx.x & 31;
    const int wave = threadIdx.x >> 5;
    const int tile = blockIdx.x * 8 + wave;
    if (tile * 16 >= rows) return;                 // wave-uniform: EXEC all-ones at WMMA
    const int row0 = tile * 16;
    const int mr = lane & 15;
    const int hi = lane >> 4;

    v8f acc[8] = {};
    const float*  xrf = (const float*)Xv  + (size_t)(row0 + mr) * HID;
    const __bf16* xrb = (const __bf16*)Xv + (size_t)(row0 + mr) * HID;

#pragma unroll
    for (int kc = 0; kc < 4; ++kc) {
        const int ks0 = kc * 32 + hi * 8;          // A layout: lanes<16 K={0-7,16-23}, lanes>=16 K={8-15,24-31}
        v16bf A;
        if (IN_BF) {
            v8bf lo = *(const v8bf*)(xrb + ks0);
            v8bf hi8 = *(const v8bf*)(xrb + ks0 + 16);
            A = __builtin_shufflevector(lo, hi8, 0,1,2,3,4,5,6,7,8,9,10,11,12,13,14,15);
        } else {
            float a[16];
            float4 p0 = *(const float4*)(xrf + ks0);
            float4 p1 = *(const float4*)(xrf + ks0 + 4);
            float4 p2 = *(const float4*)(xrf + ks0 + 16);
            float4 p3 = *(const float4*)(xrf + ks0 + 20);
            a[0]=p0.x; a[1]=p0.y; a[2]=p0.z; a[3]=p0.w;
            a[4]=p1.x; a[5]=p1.y; a[6]=p1.z; a[7]=p1.w;
            a[8]=p2.x; a[9]=p2.y; a[10]=p2.z; a[11]=p2.w;
            a[12]=p3.x; a[13]=p3.y; a[14]=p3.z; a[15]=p3.w;
#pragma unroll
            for (int t2 = 0; t2 < 16; ++t2) A[t2] = (__bf16)a[t2];
        }

        // B layout: lanes 0-15 = col N, K={0-7,16-23} (lanes 16-31: K={8-15,24-31}), matching A
        const __bf16* wbase = Wt + (size_t)(lane & 15) * HID + kc * 32 + hi * 8;
#pragma unroll
        for (int n = 0; n < 8; ++n) {
            v8bf blo = *(const v8bf*)(wbase + (size_t)n * 16 * HID);
            v8bf bhi = *(const v8bf*)(wbase + (size_t)n * 16 * HID + 16);
            v16bf B = __builtin_shufflevector(blo, bhi, 0,1,2,3,4,5,6,7,8,9,10,11,12,13,14,15);
            acc[n] = __builtin_amdgcn_wmma_f32_16x16x32_bf16(
                false, A, false, B, (short)0, acc[n], false, false);
        }
    }

    const float inv_std = 0.9999950000374997f;     // rsqrt(1 + 1e-5)
#pragma unroll
    for (int n = 0; n < 8; ++n) {
        const int cn = n * 16 + (lane & 15);
        float bs = bias[cn];
        float g = 1.0f, bt = 0.0f;
        if (USE_BN) { g = gamma[cn] * inv_std; bt = beta[cn]; }
#pragma unroll
        for (int r = 0; r < 8; ++r) {
            float v = acc[n][r] + bs;              // C/D layout: VGPR r -> row r (+8 for hi lanes)
            if (USE_BN) v = g * v + bt;
            if (doRelu) v = v > 0.f ? v : 0.f;
            const size_t oidx = (size_t)(row0 + r + hi * 8) * HID + cn;
            if (OUT_BF) ((__bf16*)Yv)[oidx] = (__bf16)v;
            else        ((float*)Yv)[oidx] = v;
        }
    }
}

extern "C" void kernel_launch(void* const* d_in, const int* in_sizes, int n_in,
                              void* d_out, int out_size, void* d_ws, size_t ws_size,
                              hipStream_t stream) {
    (void)in_sizes; (void)n_in; (void)out_size; (void)ws_size;
    const float* x_N  = (const float*)d_in[0];
    const float* We   = (const float*)d_in[1];
    const float* be   = (const float*)d_in[2];
    const float* Wg   = (const float*)d_in[3];
    const float* bg   = (const float*)d_in[4];
    const float* eps  = (const float*)d_in[5];
    const float* gam  = (const float*)d_in[6];
    const float* bet  = (const float*)d_in[7];
    const float* Wp   = (const float*)d_in[8];
    const float* bp   = (const float*)d_in[9];
    const int* oni    = (const int*)d_in[10];
    const int* n2e    = (const int*)d_in[11];
    const int* eiN    = (const int*)d_in[12];
    const int* oei    = (const int*)d_in[13];
    const int* e2n    = (const int*)d_in[14];
    const int* eiE    = (const int*)d_in[15];
    const int* bat    = (const int*)d_in[16];

    // ---- workspace layout (256B-aligned pieces) ----
    char* ws = (char*)d_ws;
    size_t off = 0;
    __bf16* Wt = (__bf16*)(ws + off);      off += (size_t)17 * 16384 * 2;        // 557056
    float* xs[3];
    for (int i = 0; i < 3; ++i) { xs[i] = (float*)(ws + off); off += (size_t)N_NODES * HID * 4; }
    float*  edge_x = (float*)(ws + off);   off += (size_t)N_EDGES * HID * 4;
    float*  G      = (float*)(ws + off);   off += (size_t)M_INC * HID * 4;       // f32 accumulator
    __bf16* A_bf   = (__bf16*)(ws + off);  off += (size_t)M_INC * HID * 2;       // x (bf16)
    __bf16* H_bf   = (__bf16*)(ws + off);  off += (size_t)M_INC * HID * 2;       // MLP hidden (bf16)
    float*  pooled = (float*)(ws + off);   off += (size_t)BATCH_B * HID * 4;

    const int gemmBlocksN = (N_NODES / 16 + 7) / 8;
    const int gemmBlocksM = (M_INC  / 16 + 7) / 8;
    const int rowBlocksM  = ((long)M_INC * 16 + 255) / 256;
    const int edgeBlocks  = ((long)L_EDG * 16 + 255) / 256;

    // weights -> bf16 transposed
    wconv_kernel<<<(17 * 16384 + 255) / 256, 256, 0, stream>>>(We, Wg, Wt);

    // embedding: xs[0] = x_N @ We + be   (f32 in, f32 out)
    gemm128<0, 0, 0><<<gemmBlocksN, 256, 0, stream>>>(x_N, Wt, be, nullptr, nullptr,
                                                      xs[0], N_NODES, 0);

    float* node_x = xs[0];
    for (int i = 0; i < 2; ++i) {
        for (int d = 0; d < 2; ++d) {
            const int* idx   = (d == 0) ? oni : oei;
            const float* ft  = (d == 0) ? node_x : edge_x;
            const int* ei    = (d == 0) ? eiN : eiE;

            gather_to_bf<<<rowBlocksM, 256, 0, stream>>>(A_bf, ft, idx, M_INC);

            for (int l = 0; l < 2; ++l) {
                int q0 = (((i * 2 + d) * 2 + l) * 2) + 0;   // mlp linear 0
                int q1 = q0 + 1;                            // mlp linear 1
                int bnIdx = (i * 2 + d) * 2 + l;

                // G = (1+eps)*x ; G[dst] += x[src]
                ginit_kernel<<<rowBlocksM, 256, 0, stream>>>(G, A_bf, eps + bnIdx, M_INC);
                scatter_edges_bf<<<edgeBlocks, 256, 0, stream>>>(G, A_bf, ei, ei + L_EDG, L_EDG);

                // H = relu(G @ W0 + b0)                (f32 in, bf16 out)
                gemm128<0, 1, 0><<<gemmBlocksM, 256, 0, stream>>>(
                    G, Wt + (size_t)(1 + q0) * 16384, bg + (size_t)q0 * HID,
                    nullptr, nullptr, H_bf, M_INC, 1);
                // x' = [relu]( BN( H @ W1 + b1 ) )     (bf16 in, bf16 out)
                gemm128<1, 1, 1><<<gemmBlocksM, 256, 0, stream>>>(
                    H_bf, Wt + (size_t)(1 + q1) * 16384, bg + (size_t)q1 * HID,
                    gam + (size_t)bnIdx * HID, bet + (size_t)bnIdx * HID,
                    A_bf, M_INC, (l < 1) ? 1 : 0);
            }

            if (d == 0) {
                hipMemsetAsync(edge_x, 0, (size_t)N_EDGES * HID * 4, stream);
                pool_rows_bf<<<rowBlocksM, 256, 0, stream>>>(edge_x, A_bf, n2e, M_INC);
                relu_ip<<<((long)N_EDGES * HID + 255) / 256, 256, 0, stream>>>(edge_x, (long)N_EDGES * HID);
            } else {
                float* nx_out = xs[i + 1];
                hipMemsetAsync(nx_out, 0, (size_t)N_NODES * HID * 4, stream);
                pool_rows_bf<<<rowBlocksM, 256, 0, stream>>>(nx_out, A_bf, e2n, M_INC);
                relu_ip<<<((long)N_NODES * HID + 255) / 256, 256, 0, stream>>>(nx_out, (long)N_NODES * HID);
                node_x = nx_out;
            }
        }
    }

    // prediction heads
    hipMemsetAsync(d_out, 0, BATCH_B * NCLS * sizeof(float), stream);
    for (int i = 0; i < 3; ++i) {
        hipMemsetAsync(pooled, 0, (size_t)BATCH_B * HID * 4, stream);
        pool_batch<<<256, 256, 0, stream>>>(pooled, xs[i], oni, bat, M_INC);
        head_kernel<<<1, BATCH_B * NCLS, 0, stream>>>((float*)d_out, pooled,
                                                      Wp + (size_t)i * HID * NCLS, bp + (size_t)i * NCLS);
    }
}
